// MinGRU_91070486544678
// MI455X (gfx1250) — matrix operand order repaired and checked by reading, not compiled, not run on hardware
//
#include <hip/hip_runtime.h>
#include <math.h>

// Problem constants (from reference setup_inputs)
#define B_   8
#define T_   4096
#define D_   1024
#define H_   1024
#define M_   (B_ * T_)      // 32768 rows of x
#define SEG  64             // segments along T
#define SEGLEN 64           // T / SEG

// Trivial (union-safe) vector types
typedef __attribute__((ext_vector_type(16))) __bf16       v16bf;
typedef __attribute__((ext_vector_type(8)))  float        v8f;
typedef __attribute__((ext_vector_type(4)))  unsigned int u32x4;
typedef __attribute__((ext_vector_type(4)))  float        f32x4;

__device__ __forceinline__ float softplus_f(float x) {
    return fmaxf(x, 0.f) + log1pf(expf(-fabsf(x)));
}
__device__ __forceinline__ float logaddexp_f(float a, float b) {
    float m = fmaxf(a, b);
    if (m == -INFINITY) return -INFINITY;
    return m + log1pf(expf(fminf(a, b) - m));
}
// log g(x): x>=0 -> log(x+0.5) ; x<0 -> -softplus(-x) = x - log1p(exp(x))
__device__ __forceinline__ float log_g_f(float x) {
    return (x >= 0.f) ? logf(x + 0.5f) : (x - log1pf(expf(x)));
}

// ---------------- Kernel 1: convert weights f32 -> bf16 (once, 4 MB) --------
__global__ void cvt_weights(const float* __restrict__ wz, const float* __restrict__ wh,
                            unsigned short* __restrict__ wzb_u, unsigned short* __restrict__ whb_u) {
    int i = blockIdx.x * blockDim.x + threadIdx.x;   // grid sized exactly H*D
    __bf16* wzb = (__bf16*)wzb_u;
    __bf16* whb = (__bf16*)whb_u;
    wzb[i] = (__bf16)wz[i];
    whb[i] = (__bf16)wh[i];
}

// ---------------- Kernel 2: fused dual-GEMM + log-space epilogue ------------
// Block tile: 64 M x 128 N, 8 waves (2 along M x 4 along N), wave tile 32x32,
// both Z and H outputs accumulated simultaneously (A-tile reused 4x per frag).
#define LDS_STRIDE 56   // bf16 elems per row: 112 B (16B-aligned, bank-spread)

__global__ __launch_bounds__(256) void gemm_act(
    const float* __restrict__ x,
    const unsigned short* __restrict__ wzb_u, const unsigned short* __restrict__ whb_u,
    const float* __restrict__ b_z, const float* __restrict__ b_h,
    float* __restrict__ lcoef, float* __restrict__ dout)
{
    const __bf16* wzb = (const __bf16*)wzb_u;
    const __bf16* whb = (const __bf16*)whb_u;

    __shared__ __bf16 ldsA[64 * LDS_STRIDE];   // 7 KB

    const int tid    = threadIdx.x;
    const int lane   = tid & 31;
    const int wave   = tid >> 5;
    const int lane15 = lane & 15;
    const int half   = lane >> 4;
    const int wave_m = wave & 1;     // 0..1
    const int wave_n = wave >> 1;    // 0..3
    const int m_base = blockIdx.x * 64;
    const int n_base = blockIdx.y * 128;

    // cooperative A-stage mapping: each thread moves 8 consecutive f32 -> 8 bf16
    const int frow = tid >> 2;       // 0..63
    const int fseg = tid & 3;        // 0..3  (k sub-chunk of 8)

    union Frag { u32x4 u[2]; v16bf v; };

    v8f accZ[2][2], accH[2][2];
    const v8f vzero = {0.f,0.f,0.f,0.f,0.f,0.f,0.f,0.f};
    #pragma unroll
    for (int i = 0; i < 2; ++i)
        #pragma unroll
        for (int j = 0; j < 2; ++j) { accZ[i][j] = vzero; accH[i][j] = vzero; }

    for (int kk = 0; kk < D_; kk += 32) {
        // ---- stage x tile (f32 -> bf16) into LDS, fully coalesced ----
        {
            const f32x4* src = reinterpret_cast<const f32x4*>(
                x + (size_t)(m_base + frow) * D_ + kk + fseg * 8);
            f32x4 f0 = src[0], f1 = src[1];
            union { u32x4 u; __bf16 h[8]; } pk;
            pk.h[0] = (__bf16)f0.x; pk.h[1] = (__bf16)f0.y;
            pk.h[2] = (__bf16)f0.z; pk.h[3] = (__bf16)f0.w;
            pk.h[4] = (__bf16)f1.x; pk.h[5] = (__bf16)f1.y;
            pk.h[6] = (__bf16)f1.z; pk.h[7] = (__bf16)f1.w;
            *reinterpret_cast<u32x4*>(&ldsA[frow * LDS_STRIDE + fseg * 8]) = pk.u;
            if (kk + 32 < D_)  // emits global_prefetch_b8 for next K-slice
                __builtin_prefetch(x + (size_t)(m_base + frow) * D_ + kk + 32 + fseg * 8, 0, 0);
        }
        __syncthreads();

        // ---- A fragments from LDS (ISA 16-bit A 16x32 layout) ----
        Frag aF[2];
        #pragma unroll
        for (int i = 0; i < 2; ++i) {
            int m_local = wave_m * 32 + i * 16 + lane15;
            const u32x4* ap = reinterpret_cast<const u32x4*>(
                &ldsA[m_local * LDS_STRIDE + half * 8]);
            aF[i].u[0] = ap[0];   // K = half*8   .. +7
            aF[i].u[1] = ap[2];   // K = 16+half*8 .. +7
        }
        // ---- B fragments direct from global (weights L2-resident) ----
        Frag bZ[2], bH[2];
        #pragma unroll
        for (int j = 0; j < 2; ++j) {
            int n = n_base + wave_n * 32 + j * 16 + lane15;
            const u32x4* pz = reinterpret_cast<const u32x4*>(
                wzb + (size_t)n * D_ + kk + half * 8);
            const u32x4* ph = reinterpret_cast<const u32x4*>(
                whb + (size_t)n * D_ + kk + half * 8);
            bZ[j].u[0] = pz[0]; bZ[j].u[1] = pz[2];
            bH[j].u[0] = ph[0]; bH[j].u[1] = ph[2];
        }
        // ---- 8 WMMAs per K-step (2 M-sub x 2 N-sub x 2 outputs) ----
        #pragma unroll
        for (int i = 0; i < 2; ++i)
            #pragma unroll
            for (int j = 0; j < 2; ++j) {
                accZ[i][j] = __builtin_amdgcn_wmma_f32_16x16x32_bf16(
                    false, aF[i].v, false, bZ[j].v, (short)0, accZ[i][j], false, false);
                accH[i][j] = __builtin_amdgcn_wmma_f32_16x16x32_bf16(
                    false, aF[i].v, false, bH[j].v, (short)0, accH[i][j], false, false);
            }
        __syncthreads();
    }

    // ---- fused epilogue: bias + log-space activations ----
    // C/D layout: element r of acc is (m = r + 8*half, n = lane15) within subtile
    #pragma unroll
    for (int j = 0; j < 2; ++j) {
        int gn = n_base + wave_n * 32 + j * 16 + lane15;
        float bz = b_z[gn], bh = b_h[gn];
        #pragma unroll
        for (int i = 0; i < 2; ++i) {
            #pragma unroll
            for (int r = 0; r < 8; ++r) {
                int gm = m_base + wave_m * 32 + i * 16 + r + 8 * half;
                float kz = accZ[i][j][r] + bz;       // k  = x·W_z + b_z
                float hp = accH[i][j][r] + bh;       // pre-activation of h~
                float sp = softplus_f(kz);
                float lc = -sp;                       // log(1 - sigmoid(k))
                float lv = (kz - sp) + log_g_f(hp);   // log sigmoid(k) + log g(.)
                lcoef[(size_t)gm * H_ + gn] = lc;
                int b = gm >> 12;                     // T_ == 4096
                int t = gm & (T_ - 1);
                dout[((size_t)b * (T_ + 1) + t + 1) * H_ + gn] = lv;  // staged log_values
            }
        }
    }
}

// ---------------- Kernel 3: per-segment reduce (A = sum lc, V = seg-lse) ----
__global__ void seg_reduce(const float* __restrict__ lcoef, const float* __restrict__ dout,
                           float* __restrict__ segA, float* __restrict__ segV) {
    int gid = blockIdx.x * blockDim.x + threadIdx.x;  // (b*SEG + s)*H + h
    int h = gid & (H_ - 1);
    int rest = gid >> 10;
    int s = rest & (SEG - 1);
    int b = rest >> 6;
    size_t lcbase = ((size_t)b * T_ + s * SEGLEN) * H_ + h;
    size_t lvbase = ((size_t)b * (T_ + 1) + s * SEGLEN + 1) * H_ + h;
    float A = 0.f, V = -INFINITY;
    for (int t = 0; t < SEGLEN; ++t) {
        float lc = lcoef[lcbase + (size_t)t * H_];
        float lv = dout [lvbase + (size_t)t * H_];
        V = logaddexp_f(V + lc, lv);
        A += lc;
    }
    segA[gid] = A;
    segV[gid] = V;
}

// ---------------- Kernel 4: scan over segments per channel ------------------
__global__ void seg_scan(const float* __restrict__ h0,
                         const float* __restrict__ segA, const float* __restrict__ segV,
                         float* __restrict__ states, float* __restrict__ dout) {
    int gid = blockIdx.x * blockDim.x + threadIdx.x;  // b*H + h
    int h = gid & (H_ - 1);
    int b = gid >> 10;
    float st = log_g_f(h0[(size_t)b * H_ + h]);
    dout[(size_t)b * (T_ + 1) * H_ + h] = expf(st);   // h[:,0,:] = g(h_0)
    for (int s = 0; s < SEG; ++s) {
        size_t idx = ((size_t)b * SEG + s) * H_ + h;
        states[idx] = st;
        st = logaddexp_f(segA[idx] + st, segV[idx]);
    }
}

// ---------------- Kernel 5: apply scan, write h = exp(state) in place -------
__global__ void seg_apply(const float* __restrict__ lcoef, const float* __restrict__ states,
                          float* __restrict__ dout) {
    int gid = blockIdx.x * blockDim.x + threadIdx.x;  // (b*SEG + s)*H + h
    int h = gid & (H_ - 1);
    int rest = gid >> 10;
    int s = rest & (SEG - 1);
    int b = rest >> 6;
    float st = states[gid];
    size_t lcbase = ((size_t)b * T_ + s * SEGLEN) * H_ + h;
    size_t lvbase = ((size_t)b * (T_ + 1) + s * SEGLEN + 1) * H_ + h;
    for (int t = 0; t < SEGLEN; ++t) {
        float lc = lcoef[lcbase + (size_t)t * H_];
        size_t oi = lvbase + (size_t)t * H_;
        float lv = dout[oi];
        st = logaddexp_f(st + lc, lv);
        dout[oi] = expf(st);
    }
}

extern "C" void kernel_launch(void* const* d_in, const int* in_sizes, int n_in,
                              void* d_out, int out_size, void* d_ws, size_t ws_size,
                              hipStream_t stream) {
    const float* x   = (const float*)d_in[0];
    const float* h0  = (const float*)d_in[1];
    const float* W_z = (const float*)d_in[2];
    const float* b_z = (const float*)d_in[3];
    const float* W_h = (const float*)d_in[4];
    const float* b_h = (const float*)d_in[5];
    float* dout = (float*)d_out;

    // workspace layout (~138 MB): bf16 weights, log_coeffs, segment state
    char* ws = (char*)d_ws;
    const size_t NK = (size_t)H_ * D_;
    unsigned short* wzb = (unsigned short*)ws;  ws += NK * 2;
    unsigned short* whb = (unsigned short*)ws;  ws += NK * 2;
    float* lcoef  = (float*)ws;                 ws += (size_t)M_ * H_ * 4;
    float* segA   = (float*)ws;                 ws += (size_t)B_ * SEG * H_ * 4;
    float* segV   = (float*)ws;                 ws += (size_t)B_ * SEG * H_ * 4;
    float* states = (float*)ws;

    cvt_weights<<<(int)(NK / 256), 256, 0, stream>>>(W_z, W_h, wzb, whb);

    dim3 grid_gemm(M_ / 64, H_ / 128);   // 512 x 8 blocks, 256 thr (8 waves)
    gemm_act<<<grid_gemm, 256, 0, stream>>>(x, wzb, whb, b_z, b_h, lcoef, dout);

    const int nseg_items = B_ * SEG * H_;  // 524288-way parallel scan phases
    seg_reduce<<<nseg_items / 256, 256, 0, stream>>>(lcoef, dout, segA, segV);
    seg_scan  <<<(B_ * H_) / 256, 256, 0, stream>>>(h0, segA, segV, states, dout);
    seg_apply <<<nseg_items / 256, 256, 0, stream>>>(lcoef, states, dout);
}
